// SetFeatureAbstraction_82248623719064
// MI455X (gfx1250) — matrix operand order repaired and checked by reading, not compiled, not run on hardware
//
#include <hip/hip_runtime.h>
#include <hip/hip_bf16.h>
#include <float.h>

#define BATCH 4
#define NPTS  4096
#define FEAT  64
#define DOUT  128
#define MCENT 2048          // NPTS/2
#define K_NN  16
#define CIN   69            // FEAT + 5
#define CPAD  96            // K padded to 3 * 32 WMMA steps (col 69 = 1.0 carries bias)
#define CROW  104           // LDS row stride in halves: 52 dwords -> conflict-free b128
#define BN_EPS 1e-5f

typedef __attribute__((ext_vector_type(16))) _Float16 v16h;
typedef __attribute__((ext_vector_type(8)))  _Float16 v8h;
typedef __attribute__((ext_vector_type(8)))  float    v8f;

// ---------------------------------------------------------------------------
// Kernel 0: build WMMA-swizzled B fragments once.
// wfrag[((wave*3 + kstep)*32 + lane)*16 + e] holds W'(K, col) as f16 where
//   K = kstep*32 + e + (lane>=16 ? 16 : 0), col = wave*16 + (lane&15),
//   row 69 = bias (multiplied by the constant-1 A column), rows 70..95 = 0.
// Each lane's per-kstep fragment is 32 contiguous, 32B-aligned bytes.
// ---------------------------------------------------------------------------
__global__ __launch_bounds__(256) void wprep_kernel(const float* __restrict__ W,
                                                    const float* __restrict__ bias,
                                                    _Float16* __restrict__ wfrag) {
    for (int s = threadIdx.x; s < 8 * 3 * 32 * 16; s += 256) {
        const int e  = s & 15;
        const int l  = (s >> 4) & 31;
        const int r2 = s >> 9;            // wave*3 + kstep
        const int ks = r2 % 3;
        const int w  = r2 / 3;
        const int K  = ks * 32 + e + ((l >= 16) ? 16 : 0);
        const int col = w * 16 + (l & 15);
        float v = 0.0f;
        if (K < CIN)       v = W[K * DOUT + col];
        else if (K == CIN) v = bias[col];
        wfrag[s] = (_Float16)v;
    }
}

// ---------------------------------------------------------------------------
// Kernel 1: FPS, one workgroup per batch. Each thread register-caches its 16
// points; per step: VALU min-update, 5-step shfl_xor wave argmax, 8-entry
// cross-wave combine (2 barriers per step). Centroids written to d_out tail.
// ---------------------------------------------------------------------------
__global__ __launch_bounds__(256) void fps_kernel(const float* __restrict__ spatial,
                                                  int* __restrict__ idxs,
                                                  float* __restrict__ cent) {
    const int b = blockIdx.x;
    const int t = threadIdx.x;
    const int lane = t & 31, wv = t >> 5;
    const float* sp = spatial + (size_t)b * NPTS * 5;

    __shared__ float s_val[8];
    __shared__ int   s_idx[8];
    __shared__ int   s_last;

    float p0[16], p1[16], p2[16], p3[16], p4[16], dmin[16];
#pragma unroll
    for (int q = 0; q < 16; ++q) {
        const int j = t + 256 * q;
        p0[q] = sp[j * 5 + 0]; p1[q] = sp[j * 5 + 1]; p2[q] = sp[j * 5 + 2];
        p3[q] = sp[j * 5 + 3]; p4[q] = sp[j * 5 + 4];
        dmin[q] = FLT_MAX;
    }

    if (t == 0) {
        s_last = 0;
        idxs[b * MCENT] = 0;
#pragma unroll
        for (int c = 0; c < 5; ++c) cent[(size_t)(b * MCENT) * 5 + c] = sp[c];
    }
    __syncthreads();

    for (int step = 1; step < MCENT; ++step) {
        const int last = s_last;
        const float l0 = sp[last * 5 + 0];
        const float l1 = sp[last * 5 + 1];
        const float l2 = sp[last * 5 + 2];
        const float l3 = sp[last * 5 + 3];
        const float l4 = sp[last * 5 + 4];

        float bestv = -FLT_MAX;
        int   besti = 0;
#pragma unroll
        for (int q = 0; q < 16; ++q) {
            const float d0 = p0[q] - l0, d1 = p1[q] - l1, d2 = p2[q] - l2;
            const float d3 = p3[q] - l3, d4 = p4[q] - l4;
            const float dd = d0 * d0 + d1 * d1 + d2 * d2 + d3 * d3 + d4 * d4;
            const float nd = fminf(dmin[q], dd);
            dmin[q] = nd;
            if (nd > bestv) { bestv = nd; besti = t + 256 * q; }
        }
        // wave argmax (ties -> lowest index)
#pragma unroll
        for (int off = 16; off > 0; off >>= 1) {
            const float ov = __shfl_xor(bestv, off, 32);
            const int   oi = __shfl_xor(besti, off, 32);
            if (ov > bestv || (ov == bestv && oi < besti)) { bestv = ov; besti = oi; }
        }
        if (lane == 0) { s_val[wv] = bestv; s_idx[wv] = besti; }
        __syncthreads();
        if (t == 0) {
            float bv = s_val[0]; int bi = s_idx[0];
#pragma unroll
            for (int w = 1; w < 8; ++w) {
                if (s_val[w] > bv || (s_val[w] == bv && s_idx[w] < bi)) {
                    bv = s_val[w]; bi = s_idx[w];
                }
            }
            s_last = bi;
            idxs[b * MCENT + step] = bi;
#pragma unroll
            for (int c = 0; c < 5; ++c)
                cent[(size_t)(b * MCENT + step) * 5 + c] = sp[bi * 5 + c];
        }
        __syncthreads();
    }
}

// ---------------------------------------------------------------------------
// Kernel 2: KNN, one thread per centroid; spatial streamed via 512-point LDS
// tiles (each block's 256 centroids are in the same batch); register-resident
// sorted top-16 with a fully unrolled insertion network.
// ---------------------------------------------------------------------------
__global__ __launch_bounds__(256) void knn_kernel(const float* __restrict__ spatial,
                                                  const int* __restrict__ idxs,
                                                  const float* __restrict__ cent,
                                                  int* __restrict__ nbr) {
    const int t   = threadIdx.x;
    const int gid = blockIdx.x * 256 + t;         // exact grid: B*MCENT threads
    const int b   = gid / MCENT;
    const float* sp = spatial + (size_t)b * NPTS * 5;

    __shared__ float s_pts[512 * 5];

    const float c0 = cent[(size_t)gid * 5 + 0];
    const float c1 = cent[(size_t)gid * 5 + 1];
    const float c2 = cent[(size_t)gid * 5 + 2];
    const float c3 = cent[(size_t)gid * 5 + 3];
    const float c4 = cent[(size_t)gid * 5 + 4];
    const int   self = idxs[gid];

    float bd[K_NN];
    int   bi[K_NN];
#pragma unroll
    for (int k = 0; k < K_NN; ++k) { bd[k] = FLT_MAX; bi[k] = 0; }

    for (int tile = 0; tile < NPTS; tile += 512) {
        __syncthreads();
        for (int s = t; s < 512 * 5; s += 256) s_pts[s] = sp[tile * 5 + s];
        __syncthreads();
        for (int jj = 0; jj < 512; ++jj) {
            const int j = tile + jj;
            const float d0 = s_pts[jj * 5 + 0] - c0;
            const float d1 = s_pts[jj * 5 + 1] - c1;
            const float d2 = s_pts[jj * 5 + 2] - c2;
            const float d3 = s_pts[jj * 5 + 3] - c3;
            const float d4 = s_pts[jj * 5 + 4] - c4;
            float dd = d0 * d0 + d1 * d1 + d2 * d2 + d3 * d3 + d4 * d4;
            if (j == self) dd = FLT_MAX;          // reference masks self with inf
            if (dd < bd[K_NN - 1]) {
#pragma unroll
                for (int k = K_NN - 1; k >= 0; --k) {
                    const float pk = (k > 0) ? bd[k - 1] : -FLT_MAX;
                    const int   pi = (k > 0) ? bi[k - 1] : 0;
                    if (dd < bd[k]) {
                        if (k == 0 || pk <= dd) { bd[k] = dd; bi[k] = j; }
                        else                    { bd[k] = pk; bi[k] = pi; }
                    }
                }
            }
        }
    }
#pragma unroll
    for (int k = 0; k < K_NN; ++k) nbr[(size_t)gid * K_NN + k] = bi[k];
}

// ---------------------------------------------------------------------------
// Shared WMMA tile: one workgroup per (b, i). A-tile = 16 neighbor rows x
// [64 feats | 5 rel coords | 1.0 bias col | 0 pad] as f16 in LDS (row stride
// 104 halves -> bank-conflict-free b128). B fragments come precomputed from
// wfrag: one contiguous 32B global load per lane per K-step (L2-resident).
// ---------------------------------------------------------------------------
__device__ __forceinline__ v8f compute_tile(const float* __restrict__ x,
                                            const float* __restrict__ spatial,
                                            const float* __restrict__ cent,
                                            const int*   __restrict__ nbr,
                                            const _Float16* __restrict__ wfrag,
                                            _Float16* sA, int* s_nbr,
                                            int b, int i) {
    const int t = threadIdx.x;
    const size_t base = (size_t)b * MCENT + i;

    if (t < K_NN) s_nbr[t] = nbr[base * K_NN + t];
    __syncthreads();

    for (int s = t; s < 16 * CROW; s += 256) {
        const int k = s / CROW, c = s % CROW;
        const int j = s_nbr[k];
        float v = 0.0f;
        if (c < FEAT) {
            v = x[((size_t)b * NPTS + j) * FEAT + c];
        } else if (c < CIN) {
            const int cc = c - FEAT;
            v = spatial[((size_t)b * NPTS + j) * 5 + cc] - cent[base * 5 + cc];
        } else if (c == CIN) {
            v = 1.0f;                     // bias column
        }
        sA[s] = (_Float16)v;
    }
    __syncthreads();

    const int lane = t & 31;
    const int wave = t >> 5;              // N-tile
    const int hi   = lane >> 4;
    const int row  = lane & 15;

    v8f acc = {};
#pragma unroll
    for (int ks = 0; ks < 3; ++ks) {
        const int kb = ks * 32;
        // A (16x32 f16): lanes 0-15 K = 0..15, lanes 16-31 K = 16..31
        const v8h alo = *(const v8h*)(sA + row * CROW + kb + hi * 16);
        const v8h ahi = *(const v8h*)(sA + row * CROW + kb + hi * 16 + 8);
        const v16h a = __builtin_shufflevector(alo, ahi,
                            0, 1, 2, 3, 4, 5, 6, 7, 8, 9, 10, 11, 12, 13, 14, 15);
        const v16h bf = *(const v16h*)(wfrag + (((wave * 3 + ks) * 32 + lane) << 4));
        acc = __builtin_amdgcn_wmma_f32_16x16x32_f16(
                  false, a, false, bf, (short)0, acc, false, false);
    }
    return acc;
}

// ---------------------------------------------------------------------------
// Kernel 3: WMMA pass 1 -> per-(b,i) partial (sum, sumsq) over 16x128 tile.
// ---------------------------------------------------------------------------
__global__ __launch_bounds__(256) void gemm_stats_kernel(const float* __restrict__ x,
                                                         const float* __restrict__ spatial,
                                                         const float* __restrict__ cent,
                                                         const int*   __restrict__ nbr,
                                                         const _Float16* __restrict__ wfrag,
                                                         float2* __restrict__ partial) {
    __shared__ _Float16 sA[16 * CROW];
    __shared__ int      s_nbr[K_NN];
    __shared__ float    rs[16], rs2[16];

    const int blk = blockIdx.x;
    const int b = blk / MCENT, i = blk % MCENT;
    const int t = threadIdx.x;
    const int lane = t & 31, wv = t >> 5;

    v8f acc = compute_tile(x, spatial, cent, nbr, wfrag, sA, s_nbr, b, i);

    float s = 0.0f, s2 = 0.0f;
#pragma unroll
    for (int r = 0; r < 8; ++r) { const float v = acc[r]; s += v; s2 += v * v; }
#pragma unroll
    for (int off = 16; off > 0; off >>= 1) {
        s  += __shfl_xor(s,  off, 32);
        s2 += __shfl_xor(s2, off, 32);
    }
    if (lane == 0) { rs[wv] = s; rs2[wv] = s2; }
    __syncthreads();
    if (t == 0) {
        float a = 0.0f, a2 = 0.0f;
#pragma unroll
        for (int w = 0; w < 8; ++w) { a += rs[w]; a2 += rs2[w]; }
        partial[blk] = make_float2(a, a2);
    }
}

// ---------------------------------------------------------------------------
// Kernel 4: fold batch partials into per-channel BN scale/shift.
// ---------------------------------------------------------------------------
__global__ __launch_bounds__(256) void stats_kernel(const float2* __restrict__ partial,
                                                    const float* __restrict__ gamma,
                                                    const float* __restrict__ beta,
                                                    float2* __restrict__ scaleshift) {
    const int i = blockIdx.x * blockDim.x + threadIdx.x;
    if (i >= MCENT) return;
    float s = 0.0f, s2 = 0.0f;
#pragma unroll
    for (int b = 0; b < BATCH; ++b) {
        const float2 p = partial[b * MCENT + i];
        s += p.x; s2 += p.y;
    }
    const float invN = 1.0f / (float)(BATCH * K_NN * DOUT);
    const float mu  = s * invN;
    const float var = s2 * invN - mu * mu;
    const float inv = rsqrtf(var + BN_EPS);
    const float sc  = inv * gamma[i];
    scaleshift[i] = make_float2(sc, beta[i] - mu * sc);
}

// ---------------------------------------------------------------------------
// Kernel 5: WMMA pass 2 -> BN affine, max over K (shfl_xor 16 pairs the two
// half-rows of each column), ReLU, store max_feature.
// ---------------------------------------------------------------------------
__global__ __launch_bounds__(256) void gemm_out_kernel(const float* __restrict__ x,
                                                       const float* __restrict__ spatial,
                                                       const float* __restrict__ cent,
                                                       const int*   __restrict__ nbr,
                                                       const _Float16* __restrict__ wfrag,
                                                       const float2* __restrict__ scaleshift,
                                                       float* __restrict__ out) {
    __shared__ _Float16 sA[16 * CROW];
    __shared__ int      s_nbr[K_NN];

    const int blk = blockIdx.x;
    const int b = blk / MCENT, i = blk % MCENT;
    const int t = threadIdx.x;

    v8f acc = compute_tile(x, spatial, cent, nbr, wfrag, sA, s_nbr, b, i);

    const float2 ss = scaleshift[i];
    float mx = -FLT_MAX;
#pragma unroll
    for (int r = 0; r < 8; ++r) mx = fmaxf(mx, acc[r] * ss.x + ss.y);  // rows 0..7 (+8)
    mx = fmaxf(mx, __shfl_xor(mx, 16, 32));                             // other half rows
    mx = fmaxf(mx, 0.0f);                                               // ReLU(max)==max(ReLU)

    const int lane = t & 31;
    if (lane < 16) {
        const int d = (t >> 5) * 16 + lane;
        out[((size_t)b * MCENT + i) * DOUT + d] = mx;
    }
}

// ---------------------------------------------------------------------------
extern "C" void kernel_launch(void* const* d_in, const int* in_sizes, int n_in,
                              void* d_out, int out_size, void* d_ws, size_t ws_size,
                              hipStream_t stream) {
    const float* x       = (const float*)d_in[0];   // [B, n, F]
    const float* spatial = (const float*)d_in[1];   // [B, n, 5]
    const float* W       = (const float*)d_in[2];   // [F+5, D]
    const float* bias    = (const float*)d_in[3];   // [D]
    const float* gamma   = (const float*)d_in[4];   // [m]
    const float* beta    = (const float*)d_in[5];   // [m]

    float* out  = (float*)d_out;                          // max_feature [B*m*D]
    float* cent = out + (size_t)BATCH * MCENT * DOUT;     // cent        [B*m*5]

    // workspace layout (~648 KB)
    int*      idxs       = (int*)d_ws;                                  // B*m
    int*      nbr        = idxs + BATCH * MCENT;                        // B*m*K
    float2*   partial    = (float2*)(nbr + BATCH * MCENT * K_NN);       // B*m
    float2*   scaleshift = partial + BATCH * MCENT;                     // m
    _Float16* wfrag      = (_Float16*)(scaleshift + MCENT);             // 8*3*32*16

    wprep_kernel<<<1, 256, 0, stream>>>(W, bias, wfrag);
    fps_kernel<<<BATCH, 256, 0, stream>>>(spatial, idxs, cent);
    knn_kernel<<<(BATCH * MCENT) / 256, 256, 0, stream>>>(spatial, idxs, cent, nbr);
    gemm_stats_kernel<<<BATCH * MCENT, 256, 0, stream>>>(x, spatial, cent, nbr, wfrag, partial);
    stats_kernel<<<(MCENT + 255) / 256, 256, 0, stream>>>(partial, gamma, beta, scaleshift);
    gemm_out_kernel<<<BATCH * MCENT, 256, 0, stream>>>(x, spatial, cent, nbr, wfrag, scaleshift, out);
}
